// S4DWrapper_41394894799295
// MI455X (gfx1250) — matrix operand order repaired and checked by reading, not compiled, not run on hardware
//
#include <hip/hip_runtime.h>
#include <hip/hip_fp16.h>
#include <hip/hip_bf16.h>

// ---------------------------------------------------------------------------
// S4D block for MI455X (gfx1250, wave32, WMMA 16x16x32 f16).
//   B=16, L=2048, H=512, N=32
// Conv = per-channel Toeplitz GEMM Y_h (L x 16) = T_h (L x L) @ U_h (L x 16),
// B=16 rides the WMMA N dimension. Toeplitz A-fragments are contiguous
// ds_load_b128 from 8 pre-shifted LDS copies of karrR=[rev(k0),k1].
// 4 row-blocks/wave share A loads (Toeplitz shift) and the B fragment:
// 5 A-loads + 2 B-loads per 4 WMMAs. The k-loop is 2x unrolled with
// ping-pong register buffers (no cur=next moves) so loads for stage i+1
// issue under the WMMAs of stage i (s_wait_dscnt <= 7, never 0).
// ---------------------------------------------------------------------------

typedef _Float16 v16h __attribute__((ext_vector_type(16)));
typedef _Float16 h8   __attribute__((ext_vector_type(8)));
typedef float    v8f  __attribute__((ext_vector_type(8)));

#define BB 16
#define LL 2048
#define HH 512
#define NN 32
#define ML (BB * LL)        // 32768 rows for output GEMM
#define H2 (2 * HH)         // 1024

static __device__ __forceinline__ v16h cat16(h8 lo, h8 hi) {
  return __builtin_shufflevector(lo, hi, 0, 1, 2, 3, 4, 5, 6, 7,
                                         8, 9, 10, 11, 12, 13, 14, 15);
}

// ---------------- lengths[b] = sum_s input_mask[b,s] ------------------------
__global__ void k_len(const float* __restrict__ m, float* __restrict__ lens) {
  int b = threadIdx.x;
  if (b < BB) {
    float s = 0.f;
    for (int i = 0; i < LL; ++i) s += m[b * LL + i];
    lens[b] = s;
  }
}

// ------------- LayerNorm + mask, transpose to ubuf[h][b][l] (f16) -----------
__global__ __launch_bounds__(256)
void k_ln(const float* __restrict__ x, const float* __restrict__ lens,
          const float* __restrict__ gamma, const float* __restrict__ beta,
          _Float16* __restrict__ ubuf) {
  int b = blockIdx.x >> 11;
  int l = blockIdx.x & (LL - 1);
  int t = threadIdx.x;
  const float* row = x + (size_t)blockIdx.x * HH;
  float v0 = row[t], v1 = row[t + 256];
  __shared__ float red[256];
  red[t] = v0 + v1;
  __syncthreads();
  for (int s = 128; s > 0; s >>= 1) {
    if (t < s) red[t] += red[t + s];
    __syncthreads();
  }
  float mu = red[0] * (1.f / (float)HH);
  __syncthreads();
  float d0 = v0 - mu, d1 = v1 - mu;
  red[t] = d0 * d0 + d1 * d1;
  __syncthreads();
  for (int s = 128; s > 0; s >>= 1) {
    if (t < s) red[t] += red[t + s];
    __syncthreads();
  }
  float rstd = rsqrtf(red[0] * (1.f / (float)HH) + 1e-5f);
  float msk = ((float)l < lens[b]) ? 1.f : 0.f;
  float s0 = (gamma[t] * d0 * rstd + beta[t]) * msk;
  float s1 = (gamma[t + 256] * d1 * rstd + beta[t + 256]) * msk;
  size_t base = (size_t)b * LL + l;
  ubuf[(size_t)t * (BB * LL) + base]         = (_Float16)s0;
  ubuf[(size_t)(t + 256) * (BB * LL) + base] = (_Float16)s1;
}

// ----- per-(h,n) discretized mode params: logr, theta, 2*Ceff (c=0,1) -------
__global__ __launch_bounds__(256)
void k_modes(const float* __restrict__ log_dt, const float* __restrict__ log_A_real,
             const float* __restrict__ A_imag, const float* __restrict__ B_re,
             const float* __restrict__ B_im, const float* __restrict__ C_re,
             const float* __restrict__ C_im, float* __restrict__ modes) {
  int idx = blockIdx.x * 256 + threadIdx.x;
  if (idx >= HH * NN) return;
  int h = idx >> 5;
  float dt = __expf(log_dt[h]);
  float ar = -__expf(log_A_real[idx]);
  float ai = A_imag[idx];
  float har = 0.5f * dt * ar, hai = 0.5f * dt * ai;   // dtA/2
  float dr = 1.f - har, di = -hai;                    // denom = 1 - dtA/2
  float inv = 1.f / (dr * dr + di * di);
  float nr = 1.f + har, ni = hai;
  // dA = num * conj(den) / |den|^2
  float dAr = (nr * dr + ni * di) * inv;
  float dAi = (ni * dr - nr * di) * inv;
  float logr = 0.5f * __logf(dAr * dAr + dAi * dAi);
  float th = atan2f(dAi, dAr);
  float br = B_re[idx], bi = B_im[idx];
  const int M = HH * NN;
  modes[idx] = logr;
  modes[M + idx] = th;
  for (int c = 0; c < 2; ++c) {
    float cr = C_re[c * M + idx], ci = C_im[c * M + idx];
    float Cr = cr * br - ci * bi;
    float Ci = cr * bi + ci * br;
    // 2 * dt * C * conj(den) / |den|^2
    float er = 2.f * dt * (Cr * dr + Ci * di) * inv;
    float ei = 2.f * dt * (Ci * dr - Cr * di) * inv;
    modes[(2 + 2 * c) * M + idx] = er;
    modes[(3 + 2 * c) * M + idx] = ei;
  }
}

// --------- karrR[h][t]: t<L -> k0[L-1-t], t>=L -> k1[t-L]  (f16) ------------
__global__ __launch_bounds__(256)
void k_kbuild(const float* __restrict__ modes, _Float16* __restrict__ karr) {
  int idx = blockIdx.x * 256 + threadIdx.x;   // over H*L
  int h = idx >> 11;
  int j = idx & (LL - 1);
  const int M = HH * NN;
  const float* lr  = modes;
  const float* th  = modes + M;
  const float* c0r = modes + 2 * M;
  const float* c0i = modes + 3 * M;
  const float* c1r = modes + 4 * M;
  const float* c1i = modes + 5 * M;
  float k0 = 0.f, k1 = 0.f;
  float fj = (float)j;
  int mb = h * NN;
#pragma unroll 4
  for (int n = 0; n < NN; ++n) {
    int m = mb + n;
    float mag = __expf(lr[m] * fj);
    float sa, ca;
    __sincosf(th[m] * fj, &sa, &ca);
    k0 += mag * (c0r[m] * ca - c0i[m] * sa);
    k1 += mag * (c1r[m] * ca - c1i[m] * sa);
  }
  size_t base = (size_t)h * (2 * LL);
  karr[base + (LL - 1 - j)] = (_Float16)k0;
  karr[base + LL + j]       = (_Float16)k1;
}

// -------- conv: per-h Toeplitz WMMA GEMM + D-skip + exact GELU --------------
// LDS: Ut[16][2048] f16 (64KB) + 8 shifted copies of karrR (64KB) = 128KB
__global__ __launch_bounds__(256)
void k_conv(const _Float16* __restrict__ ubuf, const _Float16* __restrict__ karr,
            const float* __restrict__ D, _Float16* __restrict__ yact) {
  extern __shared__ _Float16 smem[];
  _Float16* Ut   = smem;                 // 16*2048
  _Float16* kksh = smem + BB * LL;       // 8*4096
  int h = blockIdx.x;
  int tid = threadIdx.x;

  // cooperative load of U_h (16x2048 f16, transposed [b][l]) via b128
  {
    const uint4* src = (const uint4*)(ubuf + (size_t)h * (BB * LL));
    uint4* dst = (uint4*)Ut;
    for (int i = tid; i < (BB * LL) / 8; i += 256) dst[i] = src[i];
  }
  // copy karrR (shift 0), then build shifted copies 1..7
  {
    const uint4* src = (const uint4*)(karr + (size_t)h * (2 * LL));
    uint4* dst = (uint4*)kksh;
    for (int i = tid; i < (2 * LL) / 8; i += 256) dst[i] = src[i];
  }
  __syncthreads();
  for (int s = 1; s < 8; ++s)
    for (int i = tid; i < 2 * LL; i += 256)
      kksh[s * (2 * LL) + i] = (i + s < 2 * LL) ? kksh[i + s] : (_Float16)0;
  __syncthreads();

  int w = tid >> 5;
  int lane = tid & 31;
  int half = lane >> 4;
  int mrow = lane & 15;
  float Dh = D[h];

  const _Float16* ubase = Ut + (size_t)mrow * LL + half * 8;   // B col = b

  // 32 quads of row-blocks (4 x 16 rows each); 4 quads per wave
  for (int rq = w; rq < LL / 64; rq += 8) {
    int row0 = rq * 64 + mrow;                 // row of sub-block r=0
    int s = (LL - 1 - row0) & 7;               // same for all 4 sub-blocks
    // base pointer at A-load L[0] (element offset abase0 - 48), 8-aligned
    const _Float16* kp = kksh + s * (2 * LL) +
                         ((LL - 1 - row0) - s) + (half ? 8 : 0) - 48;

    v8f acc[4];
#pragma unroll
    for (int r = 0; r < 4; ++r)
      acc[r] = (v8f){0.f, 0.f, 0.f, 0.f, 0.f, 0.f, 0.f, 0.f};

    // ping-pong register buffers: no copies, loads for stage i+1 issue
    // under stage i WMMAs
    h8 A0[5], B0[2], A1[5], B1[2];
#pragma unroll
    for (int i = 0; i < 5; ++i) A0[i] = *(const h8*)(kp + 16 * i);
    B0[0] = *(const h8*)(ubase);
    B0[1] = *(const h8*)(ubase + 16);

    for (int kk = 0; kk < LL; kk += 64) {
      // prefetch stage 1 (kk+32)
#pragma unroll
      for (int i = 0; i < 5; ++i) A1[i] = *(const h8*)(kp + kk + 32 + 16 * i);
      B1[0] = *(const h8*)(ubase + kk + 32);
      B1[1] = *(const h8*)(ubase + kk + 48);
      {
        v16h Bf = cat16(B0[0], B0[1]);
#pragma unroll
        for (int r = 0; r < 4; ++r) {
          v16h Af = cat16(A0[3 - r], A0[4 - r]);   // Toeplitz shift sharing
          acc[r] = __builtin_amdgcn_wmma_f32_16x16x32_f16(
              false, Af, false, Bf, (short)0, acc[r], false, false);
        }
      }
      // prefetch stage 0 of next iteration (kk+64, wraps harmlessly)
      int nk = (kk + 64) & (LL - 1);
#pragma unroll
      for (int i = 0; i < 5; ++i) A0[i] = *(const h8*)(kp + nk + 16 * i);
      B0[0] = *(const h8*)(ubase + nk);
      B0[1] = *(const h8*)(ubase + nk + 16);
      {
        v16h Bf = cat16(B1[0], B1[1]);
#pragma unroll
        for (int r = 0; r < 4; ++r) {
          v16h Af = cat16(A1[3 - r], A1[4 - r]);
          acc[r] = __builtin_amdgcn_wmma_f32_16x16x32_f16(
              false, Af, false, Bf, (short)0, acc[r], false, false);
        }
      }
    }

    // epilogue: y += u*D; exact GELU; scatter to yact[(b*L+l)*H + h]
#pragma unroll
    for (int r = 0; r < 4; ++r) {
#pragma unroll
      for (int j = 0; j < 8; ++j) {
        int M = half ? (j + 8) : j;
        int l = rq * 64 + r * 16 + M;
        float y = acc[r][j] + (float)Ut[(size_t)mrow * LL + l] * Dh;
        float g = 0.5f * y * (1.f + erff(y * 0.70710678118f));
        yact[(size_t)(mrow * LL + l) * HH + h] = (_Float16)g;
      }
    }
  }
}

// -------------------- W_out f32 -> f16 (same [n][k] layout) -----------------
__global__ __launch_bounds__(256)
void k_wq(const float* __restrict__ w, _Float16* __restrict__ wq) {
  int i = blockIdx.x * 256 + threadIdx.x;
  if (i < H2 * HH) wq[i] = (_Float16)w[i];
}

// ------------- output GEMM: zc = gelu_y @ W^T + b  (M=32768,N=1024,K=512) ---
__global__ __launch_bounds__(256)
void k_gemm2(const _Float16* __restrict__ yact, const _Float16* __restrict__ wq,
             const float* __restrict__ bout, _Float16* __restrict__ zc) {
  int tid = threadIdx.x;
  int w = tid >> 5;
  int lane = tid & 31;
  int half = lane >> 4;
  int mrow = lane & 15;
  int wm = w >> 1;                 // 0..3 (M subtiles of 32)
  int wn = w & 1;                  // 0..1 (N subtiles of 32)
  int mbase = blockIdx.x * 128 + wm * 32;
  int nbase = blockIdx.y * 64 + wn * 32;

  const _Float16* aptr[2];
  const _Float16* bptr[2];
#pragma unroll
  for (int r = 0; r < 2; ++r)
    aptr[r] = yact + (size_t)(mbase + r * 16 + mrow) * HH + half * 8;
#pragma unroll
  for (int c = 0; c < 2; ++c)
    bptr[c] = wq + (size_t)(nbase + c * 16 + mrow) * HH + half * 8;

  v8f acc[2][2];
#pragma unroll
  for (int r = 0; r < 2; ++r)
#pragma unroll
    for (int c = 0; c < 2; ++c)
      acc[r][c] = (v8f){0.f, 0.f, 0.f, 0.f, 0.f, 0.f, 0.f, 0.f};

  // ping-pong global fragment buffers
  h8 alo0[2], ahi0[2], blo0[2], bhi0[2];
  h8 alo1[2], ahi1[2], blo1[2], bhi1[2];
#pragma unroll
  for (int r = 0; r < 2; ++r) {
    alo0[r] = *(const h8*)(aptr[r]);
    ahi0[r] = *(const h8*)(aptr[r] + 16);
  }
#pragma unroll
  for (int c = 0; c < 2; ++c) {
    blo0[c] = *(const h8*)(bptr[c]);
    bhi0[c] = *(const h8*)(bptr[c] + 16);
  }

  for (int kb = 0; kb < HH; kb += 64) {
    // prefetch stage 1 (kb+32)
#pragma unroll
    for (int r = 0; r < 2; ++r) {
      alo1[r] = *(const h8*)(aptr[r] + kb + 32);
      ahi1[r] = *(const h8*)(aptr[r] + kb + 48);
      __builtin_prefetch(aptr[r] + kb + 64, 0, 1);   // global_prefetch_b8
    }
#pragma unroll
    for (int c = 0; c < 2; ++c) {
      blo1[c] = *(const h8*)(bptr[c] + kb + 32);
      bhi1[c] = *(const h8*)(bptr[c] + kb + 48);
    }
#pragma unroll
    for (int r = 0; r < 2; ++r) {
      v16h A = cat16(alo0[r], ahi0[r]);
#pragma unroll
      for (int c = 0; c < 2; ++c) {
        acc[r][c] = __builtin_amdgcn_wmma_f32_16x16x32_f16(
            false, A, false, cat16(blo0[c], bhi0[c]), (short)0,
            acc[r][c], false, false);
      }
    }
    // prefetch stage 0 of next iteration (kb+64, wraps harmlessly)
    int nk = (kb + 64) & (HH - 1);
#pragma unroll
    for (int r = 0; r < 2; ++r) {
      alo0[r] = *(const h8*)(aptr[r] + nk);
      ahi0[r] = *(const h8*)(aptr[r] + nk + 16);
    }
#pragma unroll
    for (int c = 0; c < 2; ++c) {
      blo0[c] = *(const h8*)(bptr[c] + nk);
      bhi0[c] = *(const h8*)(bptr[c] + nk + 16);
    }
#pragma unroll
    for (int r = 0; r < 2; ++r) {
      v16h A = cat16(alo1[r], ahi1[r]);
#pragma unroll
      for (int c = 0; c < 2; ++c) {
        acc[r][c] = __builtin_amdgcn_wmma_f32_16x16x32_f16(
            false, A, false, cat16(blo1[c], bhi1[c]), (short)0,
            acc[r][c], false, false);
      }
    }
  }

#pragma unroll
  for (int r = 0; r < 2; ++r)
#pragma unroll
    for (int c = 0; c < 2; ++c) {
      int n = nbase + c * 16 + mrow;
      float bias = bout[n];
#pragma unroll
      for (int j = 0; j < 8; ++j) {
        int m = mbase + r * 16 + (half ? (j + 8) : j);
        zc[(size_t)m * H2 + n] = (_Float16)(acc[r][c][j] + bias);
      }
    }
}

// ------------------------ GLU + residual -> d_out ---------------------------
__global__ __launch_bounds__(256)
void k_glu(const _Float16* __restrict__ zc, const float* __restrict__ x,
           float* __restrict__ out) {
  size_t idx = (size_t)blockIdx.x * 256 + threadIdx.x;   // over B*L*H
  size_t m = idx >> 9;            // / HH
  int hh = (int)(idx & (HH - 1));
  float a = (float)zc[m * H2 + hh];
  float g = (float)zc[m * H2 + HH + hh];
  float sig = 1.f / (1.f + __expf(-g));
  out[idx] = a * sig + x[idx];
}

// ---------------------------------------------------------------------------
extern "C" void kernel_launch(void* const* d_in, const int* in_sizes, int n_in,
                              void* d_out, int out_size, void* d_ws, size_t ws_size,
                              hipStream_t stream) {
  const float* x        = (const float*)d_in[0];
  const float* imask    = (const float*)d_in[1];
  const float* log_dt   = (const float*)d_in[2];
  const float* log_Ar   = (const float*)d_in[3];
  const float* A_imag   = (const float*)d_in[4];
  const float* B_re     = (const float*)d_in[5];
  const float* B_im     = (const float*)d_in[6];
  const float* C_re     = (const float*)d_in[7];
  const float* C_im     = (const float*)d_in[8];
  const float* Dsk      = (const float*)d_in[9];
  const float* W_out    = (const float*)d_in[10];
  const float* b_out    = (const float*)d_in[11];
  const float* ln_gamma = (const float*)d_in[12];
  const float* ln_beta  = (const float*)d_in[13];
  float* out = (float*)d_out;

  char* ws = (char*)d_ws;
  size_t off = 0;
  auto take = [&](size_t bytes) {
    size_t p = off;
    off = (off + bytes + 255) & ~(size_t)255;
    return p;
  };
  float*     lens  = (float*)(ws + take(BB * sizeof(float)));
  float*     modes = (float*)(ws + take((size_t)6 * HH * NN * sizeof(float)));
  _Float16*  karr  = (_Float16*)(ws + take((size_t)HH * 2 * LL * sizeof(_Float16)));
  _Float16*  ubuf  = (_Float16*)(ws + take((size_t)HH * BB * LL * sizeof(_Float16)));
  _Float16*  yact  = (_Float16*)(ws + take((size_t)ML * HH * sizeof(_Float16)));
  _Float16*  wq    = (_Float16*)(ws + take((size_t)H2 * HH * sizeof(_Float16)));
  _Float16*  zc    = (_Float16*)(ws + take((size_t)ML * H2 * sizeof(_Float16)));
  (void)ws_size; (void)in_sizes; (void)n_in; (void)out_size;

  k_len<<<1, 32, 0, stream>>>(imask, lens);
  k_ln<<<BB * LL, 256, 0, stream>>>(x, lens, ln_gamma, ln_beta, ubuf);
  k_modes<<<(HH * NN + 255) / 256, 256, 0, stream>>>(
      log_dt, log_Ar, A_imag, B_re, B_im, C_re, C_im, modes);
  k_kbuild<<<(HH * LL) / 256, 256, 0, stream>>>(modes, karr);
  k_conv<<<HH, 256, 128 * 1024, stream>>>(ubuf, karr, Dsk, yact);
  k_wq<<<(H2 * HH) / 256, 256, 0, stream>>>(W_out, wq);
  dim3 g2(ML / 128, H2 / 64);
  k_gemm2<<<g2, 256, 0, stream>>>(yact, wq, b_out, zc);
  k_glu<<<(size_t)ML * HH / 256, 256, 0, stream>>>(zc, x, out);
}
